// MaskDINOEncoder_70145405878418
// MI455X (gfx1250) — compile-verified, matrix operand order checked by reading, not run on hardware
//
#include <hip/hip_runtime.h>
#include <hip/hip_bf16.h>
#include <stdint.h>

// ---------------- static problem config ----------------
#define BATCH   2
#define LEN_Q   21760        // 256 + 1024 + 4096 + 16384
#define S_TOT   (BATCH * LEN_Q)   // 43520 rows
#define DMODEL  256
#define NHEAD   8
#define NLVL    4
#define NPNT    4
#define DHEAD   32           // DMODEL / NHEAD == wave32 lanes
#define DFF     1024

typedef __attribute__((ext_vector_type(16))) __bf16 v16bf;
typedef __attribute__((ext_vector_type(8)))  __bf16 v8bf;
typedef __attribute__((ext_vector_type(8)))  float  v8f;

// ---------------- WMMA fragment load ----------------
// 16-bit A-matrix 16x32 layout (CDNA5 ISA):
//   lane L: row = L%16, kbase = (L/16)*8
//   vector elems 0..7  -> K = kbase + 0..7   (contiguous 16B)
//   vector elems 8..15 -> K = kbase + 16..23 (contiguous 16B)
// B (32x16) uses the identical mapping applied to B^T stored [N][K].
__device__ __forceinline__ v16bf load_frag16(const __bf16* p) {
  const v8bf lo = *reinterpret_cast<const v8bf*>(p);
  const v8bf hi = *reinterpret_cast<const v8bf*>(p + 16);
  v16bf r;
#pragma unroll
  for (int i = 0; i < 8; ++i) { r[i] = lo[i]; r[i + 8] = hi[i]; }
  return r;
}

// ---------------- GEMM: C[M,N] = A[M,K] * Bt[N,K]^T + bias, opt ReLU ----------------
// Register-tiled: each wave computes a 32(M) x 64(N) tile = 2 A-frags x 4 B-frags
// -> 8 WMMAs per 32-deep K step on 12 b128 loads (~21 flops/byte at L2).
// Block = 256 threads = 8 waves arranged 4(M) x 2(N): block tile 128 x 128.
// grid = (M/128, N/128).
__global__ __launch_bounds__(256) void gemm_bf16_wmma_kernel(
    const __bf16* __restrict__ A, const __bf16* __restrict__ Bt,
    const float* __restrict__ bias, float* __restrict__ Cf,
    __bf16* __restrict__ Cb, int N, int K, int relu) {
  const int lane = threadIdx.x & 31;
  const int wave = threadIdx.x >> 5;
  const int mw = wave & 3;            // 4 waves along M, 32 rows each
  const int nw = wave >> 2;           // 2 waves along N, 64 cols each
  const int m0 = blockIdx.x * 128 + mw * 32;
  const int n0 = blockIdx.y * 128 + nw * 64;
  const int r16 = lane & 15;
  const int kb  = (lane >> 4) * 8;

  const __bf16* ap0 = A + (size_t)(m0 + r16) * K + kb;
  const __bf16* ap1 = ap0 + (size_t)16 * K;
  const __bf16* bp0 = Bt + (size_t)(n0 + r16) * K + kb;

  v8f acc[2][4];
#pragma unroll
  for (int i = 0; i < 2; ++i)
#pragma unroll
    for (int j = 0; j < 4; ++j) acc[i][j] = {};

  for (int k = 0; k < K; k += 32) {
    const v16bf a0 = load_frag16(ap0 + k);
    const v16bf a1 = load_frag16(ap1 + k);
#pragma unroll
    for (int j = 0; j < 4; ++j) {
      const v16bf bv = load_frag16(bp0 + (size_t)(j * 16) * K + k);
      acc[0][j] = __builtin_amdgcn_wmma_f32_16x16x32_bf16(
          false, a0, false, bv, (short)0, acc[0][j], false, false);
      acc[1][j] = __builtin_amdgcn_wmma_f32_16x16x32_bf16(
          false, a1, false, bv, (short)0, acc[1][j], false, false);
    }
  }

  // C/D layout: VGPR r -> M = r + 8*(lane/16), N = lane%16
  const int mfrag = m0 + (lane >> 4) * 8;
#pragma unroll
  for (int j = 0; j < 4; ++j) {
    const int nn = n0 + j * 16 + r16;
    const float badd = bias ? bias[nn] : 0.0f;
#pragma unroll
    for (int i = 0; i < 2; ++i) {
      const int mbase = mfrag + i * 16;
#pragma unroll
      for (int r = 0; r < 8; ++r) {
        float v = acc[i][j][r] + badd;
        if (relu) v = fmaxf(v, 0.0f);
        const size_t idx = (size_t)(mbase + r) * N + nn;
        if (Cf) Cf[idx] = v;
        if (Cb) Cb[idx] = (__bf16)v;
      }
    }
  }
}

// ---------------- weight transpose + f32 -> bf16 : out[n*K+k] = W[k*N+n] ----------------
__global__ __launch_bounds__(256) void wcvt_kernel(
    const float* __restrict__ W, __bf16* __restrict__ out, int K, int N) {
  const int i = blockIdx.x * 256 + threadIdx.x;
  if (i >= K * N) return;
  const int n = i / K, k = i - n * K;
  out[i] = (__bf16)W[(size_t)k * N + n];
}

// ---------------- pack: NCHW levels -> [B, LEN, D] token stream ----------------
__global__ __launch_bounds__(256) void pack_kernel(
    const float* __restrict__ s0, const float* __restrict__ s1,
    const float* __restrict__ s2, const float* __restrict__ s3,
    const float* __restrict__ p0, const float* __restrict__ p1,
    const float* __restrict__ p2, const float* __restrict__ p3,
    const float* __restrict__ le,
    float* __restrict__ src, float* __restrict__ pos, __bf16* __restrict__ srcb) {
  const int i = blockIdx.x * 256 + threadIdx.x;      // exact: S_TOT*256 threads
  const int d = i & 255;
  const int t = i >> 8;
  const int q = t % LEN_Q;
  const int b = t / LEN_Q;
  int lvl, loff;
  if (q >= 5376)      { lvl = 3; loff = q - 5376; }
  else if (q >= 1280) { lvl = 2; loff = q - 1280; }
  else if (q >= 256)  { lvl = 1; loff = q - 256;  }
  else                { lvl = 0; loff = q;        }
  const int hw = 16 << lvl;                          // square levels
  const int y = loff / hw, x = loff - y * hw;
  const float* sp; const float* pp;
  if (lvl == 0)      { sp = s0; pp = p0; }
  else if (lvl == 1) { sp = s1; pp = p1; }
  else if (lvl == 2) { sp = s2; pp = p2; }
  else               { sp = s3; pp = p3; }
  const size_t idx = (((size_t)b * DMODEL + d) * hw + y) * hw + x;
  const float sv = sp[idx];
  const float pv = pp[idx] + le[lvl * DMODEL + d];
  src[i]  = sv;
  srcb[i] = (__bf16)sv;
  pos[i]  = pv;
}

// ---------------- q = src + pos, converted to bf16 ----------------
__global__ __launch_bounds__(256) void addq_kernel(
    const float* __restrict__ a, const float* __restrict__ b, __bf16* __restrict__ o) {
  const int i = blockIdx.x * 256 + threadIdx.x;      // exact multiple
  o[i] = (__bf16)(a[i] + b[i]);
}

// ---------------- deformable-attn sampler: one wave32 per (b, q, head), lane = channel ----------------
__global__ __launch_bounds__(256) void msda_kernel(
    const float* __restrict__ value,   // [S, 256]
    const float* __restrict__ offs,    // [S, 256] = (h,l,p,2)
    const float* __restrict__ logits,  // [S, 128] = (h, l*p)
    __bf16* __restrict__ outb) {       // [S, 256]
  const int lane = threadIdx.x & 31;                 // channel d within head
  const int gid  = blockIdx.x * 8 + (threadIdx.x >> 5);
  const int h = gid & 7;
  const int t = gid >> 3;                            // b*LEN + q
  const int q = t % LEN_Q;
  const int b = t / LEN_Q;

  int lvlq, loff;
  if (q >= 5376)      { lvlq = 3; loff = q - 5376; }
  else if (q >= 1280) { lvlq = 2; loff = q - 1280; }
  else if (q >= 256)  { lvlq = 1; loff = q - 256;  }
  else                { lvlq = 0; loff = q;        }
  const int wq = 16 << lvlq;
  const float refx = ((loff % wq) + 0.5f) / (float)wq;
  const float refy = ((loff / wq) + 0.5f) / (float)wq;

  // softmax over the 16 (level, point) logits for this head (uniform per wave)
  const float* lg = logits + (size_t)t * (NHEAD * NLVL * NPNT) + h * (NLVL * NPNT);
  float w16[16];
  float mx = -3.0e38f;
#pragma unroll
  for (int i = 0; i < 16; ++i) { w16[i] = lg[i]; mx = fmaxf(mx, w16[i]); }
  float sum = 0.0f;
#pragma unroll
  for (int i = 0; i < 16; ++i) { w16[i] = __expf(w16[i] - mx); sum += w16[i]; }
  const float inv = 1.0f / sum;

  const float* orow = offs + (size_t)t * 256 + h * (NLVL * NPNT * 2);
  const int LS[4] = {0, 256, 1280, 5376};
  float acc = 0.0f;
#pragma unroll
  for (int l = 0; l < NLVL; ++l) {
    const int sz = 16 << l;
    const float fsz = (float)sz;
    const float* vb = value + ((size_t)b * LEN_Q + LS[l]) * DMODEL + h * DHEAD + lane;
#pragma unroll
    for (int p = 0; p < NPNT; ++p) {
      const float ox = orow[(l * NPNT + p) * 2 + 0];
      const float oy = orow[(l * NPNT + p) * 2 + 1];
      // loc*size - 0.5  (align_corners=False), loc = ref + off / (W,H)
      const float x = refx * fsz + ox - 0.5f;
      const float y = refy * fsz + oy - 0.5f;
      const float x0f = floorf(x), y0f = floorf(y);
      const float fx = x - x0f, fy = y - y0f;
      const int x0 = (int)x0f, y0 = (int)y0f;
      const int x1 = x0 + 1,   y1 = y0 + 1;
      const bool vx0 = (x0 >= 0) & (x0 < sz), vx1 = (x1 >= 0) & (x1 < sz);
      const bool vy0 = (y0 >= 0) & (y0 < sz), vy1 = (y1 >= 0) & (y1 < sz);
      float s = 0.0f;
      if (vx0 & vy0) s += (1.0f - fx) * (1.0f - fy) * vb[(size_t)(y0 * sz + x0) * DMODEL];
      if (vx1 & vy0) s += fx * (1.0f - fy)          * vb[(size_t)(y0 * sz + x1) * DMODEL];
      if (vx0 & vy1) s += (1.0f - fx) * fy          * vb[(size_t)(y1 * sz + x0) * DMODEL];
      if (vx1 & vy1) s += fx * fy                   * vb[(size_t)(y1 * sz + x1) * DMODEL];
      acc += (w16[l * NPNT + p] * inv) * s;
    }
  }
  outb[(size_t)t * DMODEL + h * DHEAD + lane] = (__bf16)acc;
}

// ---------------- residual + LayerNorm: one wave32 per 256-wide row ----------------
__global__ __launch_bounds__(256) void resln_kernel(
    const float* __restrict__ x, const float* __restrict__ dlt,
    const float* __restrict__ g, const float* __restrict__ beta,
    float* __restrict__ outf, __bf16* __restrict__ outb) {
  const int lane = threadIdx.x & 31;
  const int row  = blockIdx.x * 8 + (threadIdx.x >> 5);
  const size_t base = (size_t)row * DMODEL;
  float v[8];
  float s = 0.0f;
#pragma unroll
  for (int j = 0; j < 8; ++j) {
    const int c = j * 32 + lane;
    v[j] = x[base + c] + dlt[base + c];
    s += v[j];
  }
#pragma unroll
  for (int o = 16; o > 0; o >>= 1) s += __shfl_xor(s, o, 32);
  const float mu = s * (1.0f / 256.0f);
  float vs = 0.0f;
#pragma unroll
  for (int j = 0; j < 8; ++j) { const float d = v[j] - mu; vs += d * d; }
#pragma unroll
  for (int o = 16; o > 0; o >>= 1) vs += __shfl_xor(vs, o, 32);
  const float rs = rsqrtf(vs * (1.0f / 256.0f) + 1e-5f);
#pragma unroll
  for (int j = 0; j < 8; ++j) {
    const int c = j * 32 + lane;
    const float y = (v[j] - mu) * rs * g[c] + beta[c];
    outf[base + c] = y;
    outb[base + c] = (__bf16)y;
  }
}

// ---------------- host driver ----------------
extern "C" void kernel_launch(void* const* d_in, const int* in_sizes, int n_in,
                              void* d_out, int out_size, void* d_ws, size_t ws_size,
                              hipStream_t stream) {
  (void)in_sizes; (void)n_in; (void)out_size; (void)ws_size;
  const float* src_in[4] = {(const float*)d_in[0], (const float*)d_in[1],
                            (const float*)d_in[2], (const float*)d_in[3]};
  const float* pos_in[4] = {(const float*)d_in[4], (const float*)d_in[5],
                            (const float*)d_in[6], (const float*)d_in[7]};
  const float* level_embed = (const float*)d_in[8];
  const float* W_off  = (const float*)d_in[9];
  const float* b_off  = (const float*)d_in[10];
  const float* W_attn = (const float*)d_in[11];
  const float* b_attn = (const float*)d_in[12];
  const float* W_val  = (const float*)d_in[13];
  const float* b_val  = (const float*)d_in[14];
  const float* W_out  = (const float*)d_in[15];
  const float* b_out  = (const float*)d_in[16];
  const float* ln1_g  = (const float*)d_in[17];
  const float* ln1_b  = (const float*)d_in[18];
  const float* W1     = (const float*)d_in[19];
  const float* b1     = (const float*)d_in[20];
  const float* W2     = (const float*)d_in[21];
  const float* b2     = (const float*)d_in[22];
  const float* ln2_g  = (const float*)d_in[23];
  const float* ln2_b  = (const float*)d_in[24];

  // ---- workspace bump allocator (256B aligned) ----
  uint8_t* wsp = (uint8_t*)d_ws;
  auto alloc = [&](size_t bytes) -> void* {
    void* p = (void*)wsp;
    wsp += (bytes + 255) & ~(size_t)255;
    return p;
  };
  const size_t S = S_TOT;
  float*  f_src = (float*)alloc(S * DMODEL * 4);
  float*  f_pos = (float*)alloc(S * DMODEL * 4);
  float*  f_tmp = (float*)alloc(S * DMODEL * 4);
  float*  f_val = (float*)alloc(S * DMODEL * 4);
  float*  f_off = (float*)alloc(S * 256 * 4);
  float*  f_log = (float*)alloc(S * 128 * 4);
  __bf16* b_src = (__bf16*)alloc(S * DMODEL * 2);
  __bf16* b_q   = (__bf16*)alloc(S * DMODEL * 2);
  __bf16* b_att = (__bf16*)alloc(S * DMODEL * 2);
  __bf16* b_hid = (__bf16*)alloc(S * DFF * 2);
  __bf16* wv = (__bf16*)alloc((size_t)6 * 256 * 256 * 2);
  __bf16* wo = (__bf16*)alloc((size_t)6 * 256 * 256 * 2);
  __bf16* wa = (__bf16*)alloc((size_t)6 * 256 * 128 * 2);
  __bf16* wu = (__bf16*)alloc((size_t)6 * 256 * 256 * 2);
  __bf16* w1 = (__bf16*)alloc((size_t)6 * 256 * 1024 * 2);
  __bf16* w2 = (__bf16*)alloc((size_t)6 * 1024 * 256 * 2);

  // ---- convert all weights (transpose to [N][K] bf16) ----
  for (int li = 0; li < 6; ++li) {
    wcvt_kernel<<<(256 * 256) / 256, 256, 0, stream>>>(W_val + (size_t)li * 256 * 256, wv + (size_t)li * 256 * 256, 256, 256);
    wcvt_kernel<<<(256 * 256) / 256, 256, 0, stream>>>(W_off + (size_t)li * 256 * 256, wo + (size_t)li * 256 * 256, 256, 256);
    wcvt_kernel<<<(256 * 128) / 256, 256, 0, stream>>>(W_attn + (size_t)li * 256 * 128, wa + (size_t)li * 256 * 128, 256, 128);
    wcvt_kernel<<<(256 * 256) / 256, 256, 0, stream>>>(W_out + (size_t)li * 256 * 256, wu + (size_t)li * 256 * 256, 256, 256);
    wcvt_kernel<<<(256 * 1024) / 256, 256, 0, stream>>>(W1 + (size_t)li * 256 * 1024, w1 + (size_t)li * 256 * 1024, 256, 1024);
    wcvt_kernel<<<(1024 * 256) / 256, 256, 0, stream>>>(W2 + (size_t)li * 1024 * 256, w2 + (size_t)li * 1024 * 256, 1024, 256);
  }

  // ---- pack NCHW levels into token stream ----
  const int elemBlocks = (int)((S * DMODEL) / 256);   // 43520
  pack_kernel<<<elemBlocks, 256, 0, stream>>>(
      src_in[0], src_in[1], src_in[2], src_in[3],
      pos_in[0], pos_in[1], pos_in[2], pos_in[3],
      level_embed, f_src, f_pos, b_src);

  const dim3 gemmBlk(256);
  const int mBlocks = (int)(S / 128);                 // 340
  const int rowBlocks = (int)(S / 8);                 // 5440 (LN rows)
  const int waveBlocks = (int)((S * NHEAD) / 8);      // 43520 (sampler)

  for (int li = 0; li < 6; ++li) {
    // q = src + pos (bf16)
    addq_kernel<<<elemBlocks, 256, 0, stream>>>(f_src, f_pos, b_q);

    // value = src @ W_val + b_val
    gemm_bf16_wmma_kernel<<<dim3(mBlocks, 256 / 128), gemmBlk, 0, stream>>>(
        b_src, wv + (size_t)li * 256 * 256, b_val + (size_t)li * 256,
        f_val, nullptr, 256, 256, 0);
    // offsets = q @ W_off + b_off
    gemm_bf16_wmma_kernel<<<dim3(mBlocks, 256 / 128), gemmBlk, 0, stream>>>(
        b_q, wo + (size_t)li * 256 * 256, b_off + (size_t)li * 256,
        f_off, nullptr, 256, 256, 0);
    // attn logits = q @ W_attn + b_attn
    gemm_bf16_wmma_kernel<<<dim3(mBlocks, 128 / 128), gemmBlk, 0, stream>>>(
        b_q, wa + (size_t)li * 256 * 128, b_attn + (size_t)li * 128,
        f_log, nullptr, 128, 256, 0);

    // deformable sampling -> bf16 attn features
    msda_kernel<<<waveBlocks, 256, 0, stream>>>(f_val, f_off, f_log, b_att);

    // output proj
    gemm_bf16_wmma_kernel<<<dim3(mBlocks, 256 / 128), gemmBlk, 0, stream>>>(
        b_att, wu + (size_t)li * 256 * 256, b_out + (size_t)li * 256,
        f_tmp, nullptr, 256, 256, 0);
    // src = LN(src + attn)
    resln_kernel<<<rowBlocks, 256, 0, stream>>>(
        f_src, f_tmp, ln1_g + (size_t)li * 256, ln1_b + (size_t)li * 256, f_src, b_src);

    // FFN: hidden = relu(src @ W1 + b1) (bf16 only)
    gemm_bf16_wmma_kernel<<<dim3(mBlocks, 1024 / 128), gemmBlk, 0, stream>>>(
        b_src, w1 + (size_t)li * 256 * 1024, b1 + (size_t)li * 1024,
        nullptr, b_hid, 1024, 256, 1);
    // ffn_out = hidden @ W2 + b2
    gemm_bf16_wmma_kernel<<<dim3(mBlocks, 256 / 128), gemmBlk, 0, stream>>>(
        b_hid, w2 + (size_t)li * 1024 * 256, b2 + (size_t)li * 256,
        f_tmp, nullptr, 256, 1024, 0);
    // src = LN(src + ffn_out)
    resln_kernel<<<rowBlocks, 256, 0, stream>>>(
        f_src, f_tmp, ln2_g + (size_t)li * 256, ln2_b + (size_t)li * 256, f_src, b_src);
  }

  hipMemcpyAsync(d_out, f_src, S * DMODEL * sizeof(float),
                 hipMemcpyDeviceToDevice, stream);
}
